// TensorProductConvLayer_14491219657358
// MI455X (gfx1250) — matrix operand order repaired and checked by reading, not compiled, not run on hardware
//
#include <hip/hip_runtime.h>

// CDNA5 / gfx1250 fused TensorProductConvLayer.
// Edge kernel: per-16-edge-tile bf16 WMMA GEMMs (h = relu(ea@w1+b1), w = h@w2+b2)
// with the 16x1024 w-tile consumed on the fly into 5 f32 accumulators, then
// atomic scatter into agg/cnt. Node kernels: mean-aggregate + residual + BN stats
// + normalize.

typedef __attribute__((ext_vector_type(16))) __bf16 bf16x16;
typedef __attribute__((ext_vector_type(8)))  float  f32x8;

namespace {
constexpr float kInvSqrt3 = 0.57735026918962576f;
constexpr float kAlpha    = 0.17677669529663687f;

constexpr int kWaves       = 8;           // waves per block, 16 edges each
constexpr int kW1Elems     = 64 * 64;     // w1^T bf16 in LDS
constexpr int kW2ChunkN    = 512;         // half of the 1024 w2 columns per stage
constexpr int kW2Elems     = kW2ChunkN * 64;
constexpr int kHElems      = kWaves * 16 * 64;
constexpr int kCoefPerWave = 6 * 256 + 64; // css,cvv,cxs,cxv0..2 [i][m] + sh rows
}

// acc[r] += coef[i*16 + (r + 8*half)] * w[r], coefficients loaded as 2x float4
#define ACC8(ACCV, CPTR, WV) do {                               \
    const float4 _cl = *(const float4*)(CPTR);                  \
    const float4 _ch = *(const float4*)((CPTR) + 4);            \
    (ACCV)[0] += _cl.x * (WV)[0]; (ACCV)[1] += _cl.y * (WV)[1]; \
    (ACCV)[2] += _cl.z * (WV)[2]; (ACCV)[3] += _cl.w * (WV)[3]; \
    (ACCV)[4] += _ch.x * (WV)[4]; (ACCV)[5] += _ch.y * (WV)[5]; \
    (ACCV)[6] += _ch.z * (WV)[6]; (ACCV)[7] += _ch.w * (WV)[7]; \
  } while (0)

__global__ __launch_bounds__(256)
void edge_tp_kernel(const float* __restrict__ node_attr,
                    const float* __restrict__ edge_attr,
                    const float* __restrict__ edge_sh,
                    const float* __restrict__ w1,
                    const float* __restrict__ b1,
                    const float* __restrict__ w2,
                    const float* __restrict__ b2,
                    const int*   __restrict__ edge_index,
                    float* __restrict__ agg,
                    float* __restrict__ cnt,
                    int E)
{
  extern __shared__ char smem_raw[];
  __bf16* w1t  = reinterpret_cast<__bf16*>(smem_raw);   // [64 n][64 k]
  __bf16* w2t  = w1t + kW1Elems;                        // [512 nl][64 c] (chunked)
  __bf16* hbuf = w2t + kW2Elems;                        // per wave [16 m][64 c]
  float*  coef = reinterpret_cast<float*>(hbuf + kHElems);

  const int tid  = threadIdx.x;
  const int lane = tid & 31;
  const int wid  = tid >> 5;
  const int half = lane >> 4;   // lane group (rows/K-halves per WMMA layout)
  const int col  = lane & 15;

  // ---- stage w1 transposed to bf16: w1t[n*64+k] = w1[k*64+n] (coalesced reads)
#pragma unroll
  for (int it = 0; it < kW1Elems / 256; ++it) {
    const int idx = it * 256 + tid;
    const int k = idx >> 6, n = idx & 63;
    w1t[n * 64 + k] = (__bf16)w1[idx];
  }
  // ---- stage w2 chunk 0 (columns 0..511) transposed to bf16
#pragma unroll 4
  for (int it = 0; it < kW2Elems / 256; ++it) {
    const int idx = it * 256 + tid;
    const int c = idx >> 9, nl = idx & 511;
    w2t[nl * 64 + c] = (__bf16)w2[c * 1024 + nl];
  }
  __syncthreads();

  const int  te     = blockIdx.x * kWaves + wid;
  const int  e0     = te * 16;
  const bool active = e0 < E;

  float* cw   = coef + wid * kCoefPerWave;
  float* css  = cw;            // xs*ss           [i*16+m]
  float* cvv  = cw + 256;      // (xv.sv)/sqrt3   [i*16+m]
  float* cxs  = cw + 512;      // xs              [i*16+m]
  float* cxv0 = cw + 768;      // xv components   [i*16+m]
  float* cxv1 = cw + 1024;
  float* cxv2 = cw + 1280;
  float* shb  = cw + 1536;     // edge_sh rows    [m*4+c]
  __bf16* hb  = hbuf + wid * (16 * 64);

  f32x8 S  = {0.f,0.f,0.f,0.f,0.f,0.f,0.f,0.f};
  f32x8 P  = S, Q0 = S, Q1 = S, Q2 = S;
  bf16x16 a2k0, a2k1;          // h as GEMM2 A-fragments (persist across phases)

  // 16x16 tile of w for n-tile jl (within current LDS chunk); gcol0 = global col
  auto wtile = [&](int jl, int gcol0) -> f32x8 {
    f32x8 wacc = {0.f,0.f,0.f,0.f,0.f,0.f,0.f,0.f};
#pragma unroll
    for (int ks = 0; ks < 2; ++ks) {
      const __bf16* p = w2t + ((jl * 16 + col) * 64 + 16 * half + 32 * ks);
      bf16x16 b;
#pragma unroll
      for (int t = 0; t < 16; ++t) b[t] = p[t];
      wacc = __builtin_amdgcn_wmma_f32_16x16x32_bf16(
          false, ks ? a2k1 : a2k0, false, b, (short)0, wacc, false, false);
    }
    const float bb = b2[gcol0 + col];
#pragma unroll
    for (int r = 0; r < 8; ++r) wacc[r] += bb;
    return wacc;
  };

  if (active) {
    // ---- per-edge coefficient staging (lane = (i-half, edge m))
    {
      const int m = col;
      const int e = e0 + m;
      const float ssv = edge_sh[(size_t)e * 4 + 0];
      const float sv0 = edge_sh[(size_t)e * 4 + 1];
      const float sv1 = edge_sh[(size_t)e * 4 + 2];
      const float sv2 = edge_sh[(size_t)e * 4 + 3];
      if (half == 0) {
        shb[m * 4 + 0] = ssv; shb[m * 4 + 1] = sv0;
        shb[m * 4 + 2] = sv1; shb[m * 4 + 3] = sv2;
      }
      const int dst = edge_index[E + e];
      const float* x = node_attr + (size_t)dst * 64;
#pragma unroll
      for (int q = 0; q < 8; ++q) {
        const int i = half * 8 + q;
        const float xs = x[i];
        const float x0 = x[16 + i * 3 + 0];
        const float x1 = x[16 + i * 3 + 1];
        const float x2 = x[16 + i * 3 + 2];
        const int o = i * 16 + m;
        css[o]  = xs * ssv;
        cvv[o]  = kInvSqrt3 * (x0 * sv0 + x1 * sv1 + x2 * sv2);
        cxs[o]  = xs;
        cxv0[o] = x0; cxv1[o] = x1; cxv2[o] = x2;
      }
    }

    // ---- GEMM1: h = relu(edge_attr @ w1 + b1) via bf16 WMMA
    bf16x16 ea0, ea1;
    {
      const float* src = edge_attr + (size_t)(e0 + col) * 64;  // row m = col
#pragma unroll
      for (int ks = 0; ks < 2; ++ks) {
        const float* p1 = src + 32 * ks + 16 * half;           // contiguous K: k = 16*half + t
        bf16x16 a;
#pragma unroll
        for (int t = 0; t < 8; ++t) a[t] = (__bf16)p1[t];
#pragma unroll
        for (int t = 0; t < 8; ++t) a[8 + t] = (__bf16)p1[8 + t];
        if (ks == 0) ea0 = a; else ea1 = a;
      }
    }
#pragma unroll
    for (int nt = 0; nt < 4; ++nt) {
      f32x8 hacc = {0.f,0.f,0.f,0.f,0.f,0.f,0.f,0.f};
#pragma unroll
      for (int ks = 0; ks < 2; ++ks) {
        const __bf16* p = w1t + ((nt * 16 + col) * 64 + 16 * half + 32 * ks);
        bf16x16 b;
#pragma unroll
        for (int t = 0; t < 16; ++t) b[t] = p[t];
        hacc = __builtin_amdgcn_wmma_f32_16x16x32_bf16(
            false, ks ? ea1 : ea0, false, b, (short)0, hacc, false, false);
      }
      const float bb = b1[nt * 16 + col];
#pragma unroll
      for (int r = 0; r < 8; ++r) {
        float v = hacc[r] + bb;
        v = v > 0.f ? v : 0.f;
        hb[(r + 8 * half) * 64 + nt * 16 + col] = (__bf16)v;
      }
    }

    // cross-lane LDS handoff within the wave: keep compiler from reordering
    __builtin_amdgcn_wave_barrier();

    // ---- load h rows as GEMM2 A-fragments
#pragma unroll
    for (int ks = 0; ks < 2; ++ks) {
      const __bf16* p1 = hb + col * 64 + 32 * ks + 16 * half;
      bf16x16 a;
#pragma unroll
      for (int t = 0; t < 8; ++t) a[t] = p1[t];
#pragma unroll
      for (int t = 0; t < 8; ++t) a[8 + t] = p1[8 + t];
      if (ks == 0) a2k0 = a; else a2k1 = a;
    }

    // ---- contraction phase 0: t=0 (w_ss -> S), t=1 (w_vv -> S)
#pragma unroll 2
    for (int i = 0; i < 16; ++i) {
      f32x8 w = wtile(i, i * 16);
      ACC8(S, css + i * 16 + 8 * half, w);
    }
#pragma unroll 2
    for (int i = 0; i < 16; ++i) {
      f32x8 w = wtile(16 + i, (16 + i) * 16);
      ACC8(S, cvv + i * 16 + 8 * half, w);
    }
  }

  __syncthreads();
  // ---- stage w2 chunk 1 (columns 512..1023)
#pragma unroll 4
  for (int it = 0; it < kW2Elems / 256; ++it) {
    const int idx = it * 256 + tid;
    const int c = idx >> 9, nl = idx & 511;
    w2t[nl * 64 + c] = (__bf16)w2[c * 1024 + 512 + nl];
  }
  __syncthreads();

  if (active) {
    // ---- contraction phase 1: t=2 (w_sv -> P), t=3 (w_vs -> Q0..2)
#pragma unroll 2
    for (int i = 0; i < 16; ++i) {
      f32x8 w = wtile(i, (32 + i) * 16);
      ACC8(P, cxs + i * 16 + 8 * half, w);
    }
#pragma unroll 2
    for (int i = 0; i < 16; ++i) {
      f32x8 w = wtile(16 + i, (48 + i) * 16);
      ACC8(Q0, cxv0 + i * 16 + 8 * half, w);
      ACC8(Q1, cxv1 + i * 16 + 8 * half, w);
      ACC8(Q2, cxv2 + i * 16 + 8 * half, w);
    }

    // ---- combine and atomic scatter (lane holds (m = r+8*half, k = col))
    const float kAIS = kAlpha * kInvSqrt3;
#pragma unroll
    for (int r = 0; r < 8; ++r) {
      const int m = r + 8 * half;
      const int e = e0 + m;
      const float ssv = shb[m * 4 + 0];
      const float sv0 = shb[m * 4 + 1];
      const float sv1 = shb[m * 4 + 2];
      const float sv2 = shb[m * 4 + 3];
      const int src = edge_index[e];
      float* row = agg + (size_t)src * 64;
      const float outs = kAlpha * S[r];
      const float pp = kAIS * P[r];
      const float qs = kAIS * ssv;
      atomicAdd(row + col, outs);
      atomicAdd(row + 16 + col * 3 + 0, sv0 * pp + qs * Q0[r]);
      atomicAdd(row + 16 + col * 3 + 1, sv1 * pp + qs * Q1[r]);
      atomicAdd(row + 16 + col * 3 + 2, sv2 * pp + qs * Q2[r]);
      if (col == 0) atomicAdd(cnt + src, 1.0f);   // lanes 0 & 16 cover m=0..15
    }
  }
}

__global__ __launch_bounds__(256)
void node_pass1(const float* __restrict__ agg,
                const float* __restrict__ cnt,
                const float* __restrict__ node_attr,
                float* __restrict__ out,
                float* __restrict__ stats,
                int N)
{
  const int n = blockIdx.x * blockDim.x + threadIdx.x;
  float sval[16], vsq[16];
#pragma unroll
  for (int c = 0; c < 16; ++c) { sval[c] = 0.f; vsq[c] = 0.f; }
  if (n < N) {
    const float inv = 1.0f / fmaxf(cnt[n], 1.0f);
    const float* a = agg + (size_t)n * 64;
    const float* x = node_attr + (size_t)n * 64;
    float* o = out + (size_t)n * 64;
#pragma unroll
    for (int c = 0; c < 16; ++c) {
      const float v = a[c] * inv + x[c];
      o[c] = v;
      sval[c] = v;
    }
#pragma unroll
    for (int c = 0; c < 16; ++c) {
      const float v0 = a[16 + 3 * c + 0] * inv + x[16 + 3 * c + 0];
      const float v1 = a[16 + 3 * c + 1] * inv + x[16 + 3 * c + 1];
      const float v2 = a[16 + 3 * c + 2] * inv + x[16 + 3 * c + 2];
      o[16 + 3 * c + 0] = v0;
      o[16 + 3 * c + 1] = v1;
      o[16 + 3 * c + 2] = v2;
      vsq[c] = (v0 * v0 + v1 * v1 + v2 * v2) * (1.0f / 3.0f);
    }
  }
  // wave32 reduction, then one atomic per wave per channel
#pragma unroll
  for (int c = 0; c < 16; ++c) {
    float s = sval[c];
    float q = s * s;
    float v = vsq[c];
#pragma unroll
    for (int off = 16; off >= 1; off >>= 1) {
      s += __shfl_xor(s, off, 32);
      q += __shfl_xor(q, off, 32);
      v += __shfl_xor(v, off, 32);
    }
    if ((threadIdx.x & 31) == 0) {
      atomicAdd(stats + c, s);
      atomicAdd(stats + 16 + c, q);
      atomicAdd(stats + 32 + c, v);
    }
  }
}

__global__ void finalize_stats(const float* __restrict__ stats,
                               const float* __restrict__ bn_weight,
                               float* __restrict__ params,
                               int N)
{
  const int c = threadIdx.x;
  if (c < 16) {
    const float invN = 1.0f / (float)N;
    const float mean = stats[c] * invN;
    const float var  = stats[16 + c] * invN - mean * mean;
    const float fn   = stats[32 + c] * invN;
    params[c]       = mean;
    params[16 + c]  = bn_weight[c] * rsqrtf(var + 1e-5f);
    params[32 + c]  = bn_weight[16 + c] * rsqrtf(fn + 1e-5f);
  }
}

__global__ __launch_bounds__(256)
void normalize_kernel(float* __restrict__ out,
                      const float* __restrict__ params,
                      const float* __restrict__ bn_bias,
                      int total)
{
  const int idx = blockIdx.x * blockDim.x + threadIdx.x;
  if (idx >= total) return;
  const int f = idx & 63;
  const float x = out[idx];
  if (f < 16) {
    out[idx] = (x - params[f]) * params[16 + f] + bn_bias[f];
  } else {
    const int i = (f - 16) / 3;
    out[idx] = x * params[32 + i];
  }
}

extern "C" void kernel_launch(void* const* d_in, const int* in_sizes, int n_in,
                              void* d_out, int out_size, void* d_ws, size_t ws_size,
                              hipStream_t stream)
{
  const float* node_attr = (const float*)d_in[0];
  const float* edge_attr = (const float*)d_in[1];
  const float* edge_sh   = (const float*)d_in[2];
  const float* w1        = (const float*)d_in[3];
  const float* b1        = (const float*)d_in[4];
  const float* w2        = (const float*)d_in[5];
  const float* b2        = (const float*)d_in[6];
  const float* bn_weight = (const float*)d_in[7];
  const float* bn_bias   = (const float*)d_in[8];
  const int*   edge_index = (const int*)d_in[9];

  const int N = in_sizes[0] / 64;
  const int E = in_sizes[2] / 4;

  float* agg    = (float*)d_ws;
  float* cnt    = agg + (size_t)N * 64;
  float* stats  = cnt + N;
  float* params = stats + 48;
  const size_t zero_bytes = ((size_t)N * 64 + N + 48) * sizeof(float);
  hipMemsetAsync(d_ws, 0, zero_bytes, stream);

  const size_t lds_bytes = (size_t)(kW1Elems + kW2Elems + kHElems) * sizeof(__bf16)
                         + (size_t)kWaves * kCoefPerWave * sizeof(float);
  (void)hipFuncSetAttribute((const void*)edge_tp_kernel,
                            hipFuncAttributeMaxDynamicSharedMemorySize,
                            (int)lds_bytes);

  const int tiles  = (E + 15) / 16;
  const int blocks = (tiles + kWaves - 1) / kWaves;
  edge_tp_kernel<<<blocks, 256, lds_bytes, stream>>>(
      node_attr, edge_attr, edge_sh, w1, b1, w2, b2, edge_index, agg, cnt, E);

  node_pass1<<<(N + 255) / 256, 256, 0, stream>>>(agg, cnt, node_attr,
                                                  (float*)d_out, stats, N);
  finalize_stats<<<1, 32, 0, stream>>>(stats, bn_weight, params, N);

  const int total = N * 64;
  normalize_kernel<<<(total + 255) / 256, 256, 0, stream>>>(
      (float*)d_out, params, bn_bias, total);
}
